// Block_70952859730367
// MI455X (gfx1250) — matrix operand order repaired and checked by reading, not compiled, not run on hardware
//
#include <hip/hip_runtime.h>

typedef __attribute__((ext_vector_type(16))) _Float16 v16h;
typedef __attribute__((ext_vector_type(8)))  float    v8f;
typedef unsigned int u32x4 __attribute__((ext_vector_type(4)));
typedef int          i32x4 __attribute__((ext_vector_type(4)));
typedef int          i32x8 __attribute__((ext_vector_type(8)));

constexpr int KCHUNK = 512;            // K halfs staged per TDM transfer
constexpr int LPAD   = 8;              // 16B LDS pad per row (bank spread)
constexpr int LDB    = KCHUNK + LPAD;  // LDS row stride in halfs

// ---------------------------------------------------------------------------
// WMMA wrapper: D = A(16x32 f16) * B(32x16 f16) + C(16x16 f32)
// ---------------------------------------------------------------------------
__device__ __forceinline__ v8f wmma_f16(v16h a, v16h b, v8f c) {
    return __builtin_amdgcn_wmma_f32_16x16x32_f16(
        false, a, false, b, (short)0, c, false, false);
}

// ---------------------------------------------------------------------------
// Fragment loaders (wave32; contiguous k order: element s of lane l is k = 16*(l/16) + s)
// ---------------------------------------------------------------------------
__device__ __forceinline__ v16h load_a_frag(const _Float16* A, int lda,
                                            int row0, int k0) {
    const int lane = threadIdx.x & 31;
    const int sub  = lane >> 4;
    const int m    = (lane & 15) + row0;
    const _Float16* base = A + (long)m * lda + k0;
    v16h a;
#pragma unroll
    for (int p = 0; p < 8; ++p) {
        int kb = sub * 16 + 2 * p;
        a[2 * p]     = base[kb];
        a[2 * p + 1] = base[kb + 1];
    }
    return a;
}

// B 32x16 (KxN): lane n = lane&15; lanes 0-15 K=0..15, lanes 16-31 K=16..31.
// Source layout [N,K] row-major -> 32 contiguous bytes per lane.
__device__ __forceinline__ v16h load_b_frag_nk(const _Float16* Bp, int ldb,
                                               int n0, int k0) {
    const int lane = threadIdx.x & 31;
    const int sub  = lane >> 4;
    const int nn   = (lane & 15) + n0;
    const _Float16* base = Bp + (long)nn * ldb + k0 + sub * 16;
    v16h b;
#pragma unroll
    for (int p = 0; p < 8; ++p) {
        b[2 * p]     = base[2 * p];
        b[2 * p + 1] = base[2 * p + 1];
    }
    return b;
}

// ---------------------------------------------------------------------------
// Tensor Data Mover staging (CDNA5 ISA):
// 64-row x KCHUNK-half panel, data_size=8B, pad 16B per 1024B row.
// Issue / wait are split so the GEMM can double-buffer (TDM ops from one
// wave retire in order, so s_wait_tensorcnt 1 == "previous slab landed").
// ---------------------------------------------------------------------------
#if __has_builtin(__builtin_amdgcn_tensor_load_to_lds) && \
    __has_builtin(__builtin_amdgcn_s_wait_tensorcnt)
#define USE_TDM 1
#else
#define USE_TDM 0
#endif

__device__ __forceinline__ void tdm_issue(const _Float16* gsrc,
                                          _Float16* Bt, int K) {
#if USE_TDM
    if ((threadIdx.x >> 5) == 0) {  // one wave owns the DMA queue
        unsigned long long ga = (unsigned long long)(size_t)gsrc;
        unsigned ldsOff = (unsigned)(size_t)Bt;   // addr[31:0] = LDS offset
        u32x4 g0;
        g0[0] = 1u;                                // count=1, user descriptor
        g0[1] = ldsOff;                            // lds_addr
        g0[2] = (unsigned)(ga & 0xffffffffu);      // global_addr[31:0]
        g0[3] = (unsigned)((ga >> 32) & 0x1ffffffu) | (2u << 30);  // type=2
        const unsigned td0   = (unsigned)(K >> 2);       // row len, 8B units
        const unsigned tile0 = (unsigned)(KCHUNK >> 2);  // chunk len, 8B units
        i32x8 g1;
        g1[0] = (int)((3u << 16)        // data_size = 8B
                      | (1u << 20)      // pad_enable
                      | (7u << 22)      // pad_interval: every 256 DWORDs
                      | (3u << 25));    // pad_amount: 4 DWORDs (16B)
        g1[1] = (int)((td0 & 0xffffu) << 16);     // tensor_dim0 lo16
        g1[2] = (int)((td0 >> 16) | (64u << 16)); // td0 hi | tensor_dim1 lo
        g1[3] = (int)((tile0 & 0xffffu) << 16);   // tile_dim0
        g1[4] = 64;                               // tile_dim1 = 64 rows
        g1[5] = (int)td0;                         // dim0_stride lo32
        g1[6] = 0;
        g1[7] = 0;
        i32x4 z4 = {0, 0, 0, 0};
#if defined(__clang_major__) && __clang_major__ >= 23
        i32x8 z8 = {0, 0, 0, 0, 0, 0, 0, 0};
        __builtin_amdgcn_tensor_load_to_lds(g0, g1, z4, z4, z8, 0);
#else
        __builtin_amdgcn_tensor_load_to_lds(g0, g1, z4, z4, 0);
#endif
    }
#else
    // Fallback: cooperative 16B vector copy (all threads)
    const int chunksPerRow = KCHUNK / 8;
    for (int i = threadIdx.x; i < 64 * chunksPerRow; i += blockDim.x) {
        const int r = i / chunksPerRow, c = i % chunksPerRow;
        *(uint4*)(Bt + (long)r * LDB + c * 8) =
            *(const uint4*)(gsrc + (long)r * K + c * 8);
    }
#endif
}

template <int N>
__device__ __forceinline__ void tdm_wait() {
#if USE_TDM
    if ((threadIdx.x >> 5) == 0) __builtin_amdgcn_s_wait_tensorcnt(N);
#endif
}

// ---------------------------------------------------------------------------
// f32 -> f16 conversion (grid-stride)
// ---------------------------------------------------------------------------
__global__ void cvt_f32_f16(const float* __restrict__ s,
                            _Float16* __restrict__ d, long n) {
    long i = (long)blockIdx.x * blockDim.x + threadIdx.x;
    long stride = (long)gridDim.x * blockDim.x;
    for (; i < n; i += stride) d[i] = (_Float16)s[i];
}

// [Hh, R, Cc] f32  ->  [Hh, Cc, R] f16   (einsum weights -> [N,K] layout)
__global__ void cvt_t_f32_f16(const float* __restrict__ s,
                              _Float16* __restrict__ d, int R, int Cc, long n) {
    long i = (long)blockIdx.x * blockDim.x + threadIdx.x;
    long stride = (long)gridDim.x * blockDim.x;
    const long plane = (long)R * Cc;
    for (; i < n; i += stride) {
        long h = i / plane, rem = i - h * plane;
        long r = rem / Cc, c = rem - r * Cc;
        d[h * plane + c * R + r] = (_Float16)s[i];
    }
}

// ---------------------------------------------------------------------------
// WMMA GEMM:  O = act(A[M,K] * Bw[N,K]^T (+bias)) (+res)
// Block = 128 threads (4 waves); block tile 128x64, wave tile 32x64.
// B panel double-buffered in LDS via TDM: slab i+1 DMA overlaps slab i math.
// grid.z batching: Aoff=(z/divA)*sA, Boff=(z%modB)*sB, Ooff=z*sO.
// TRO=true stores O16 transposed ([N,M], used to produce V^T).
// ---------------------------------------------------------------------------
template <bool SILU, bool RES, bool TRO>
__global__ __launch_bounds__(128) void gemm16(
    const _Float16* __restrict__ A, const _Float16* __restrict__ Bw,
    const float* __restrict__ bias, const float* __restrict__ res,
    float* __restrict__ O32, _Float16* __restrict__ O16,
    int M, int N, int K, int divA, long sA, int modB, long sB, long sO) {
    __shared__ __align__(16) _Float16 Bt[2][64 * LDB];   // 2 x ~65 KB

    const int z = blockIdx.z;
    const _Float16* Ab = A + (long)(z / divA) * sA;
    const _Float16* Bb = Bw + (long)(z % modB) * sB;
    const long oOff = (long)z * sO;

    const int wave = threadIdx.x >> 5;
    const int row0 = blockIdx.y * 128 + wave * 32;
    const int col0 = blockIdx.x * 64;
    const _Float16* Bpanel = Bb + (long)col0 * K;

    v8f acc[2][4] = {};
    int buf = 0;
    tdm_issue(Bpanel, Bt[0], K);                    // prime slab 0
    for (int kb = 0; kb < K; kb += KCHUNK) {
        if (kb + KCHUNK < K) {
            tdm_issue(Bpanel + kb + KCHUNK, Bt[buf ^ 1], K);  // async next
            tdm_wait<1>();                          // slab kb has landed
        } else {
            tdm_wait<0>();
        }
        __syncthreads();                            // slab kb visible
        const _Float16* Bcur = Bt[buf];
        for (int k0 = 0; k0 < KCHUNK; k0 += 32) {
            if (kb + k0 + 32 < K)  // global_prefetch_b8
                __builtin_prefetch(Ab + (long)row0 * K + kb + k0 + 32, 0, 3);
            v16h a0 = load_a_frag(Ab, K, row0, kb + k0);
            v16h a1 = load_a_frag(Ab, K, row0 + 16, kb + k0);
#pragma unroll
            for (int t = 0; t < 4; ++t) {
                v16h b = load_b_frag_nk(Bcur, LDB, t * 16, k0);
                acc[0][t] = wmma_f16(a0, b, acc[0][t]);
                acc[1][t] = wmma_f16(a1, b, acc[1][t]);
            }
        }
        __syncthreads();                            // slab kb fully consumed
        buf ^= 1;
    }

    const int lane = threadIdx.x & 31;
    const int sub  = lane >> 4;
    const int nn   = lane & 15;
#pragma unroll
    for (int mtile = 0; mtile < 2; ++mtile)
#pragma unroll
        for (int t = 0; t < 4; ++t) {
            const int col = col0 + t * 16 + nn;
            const float bv = bias ? bias[col] : 0.0f;
#pragma unroll
            for (int r = 0; r < 8; ++r) {
                const int row = row0 + mtile * 16 + sub * 8 + r;
                const long idx = TRO ? (oOff + (long)col * M + row)
                                     : (oOff + (long)row * N + col);
                float val = acc[mtile][t][r] + bv;
                if (SILU) val = val / (1.0f + __expf(-val));
                if (RES) val += res[idx];
                if (O32) O32[idx] = val;
                if (O16) O16[idx] = (_Float16)val;
            }
        }
}

// ---------------------------------------------------------------------------
// Flash attention: one wave per (b, h, 16-query tile).
// K stored [T,HS] ([N,K] for QK^T); V stored transposed [HS,T] ([N,K] for PV).
// Row-max via 4 shuffles/row; row-SUM via one extra WMMA against an all-ones
// B matrix (P is already an A-fragment for P*V, and C-layout drops row m's
// sum at element r of every lane) -- halves the ds_bpermute/wait chains.
// ---------------------------------------------------------------------------
__global__ __launch_bounds__(32) void attn_kernel(
    const _Float16* __restrict__ q, const _Float16* __restrict__ k,
    const _Float16* __restrict__ vT, _Float16* __restrict__ attn,
    int T, int H, int HS, int C, float scale) {
    __shared__ _Float16 pbuf[16 * 32];

    const int lane = threadIdx.x & 31;
    const int sub  = lane >> 4;
    const int nn   = lane & 15;
    const int q0 = blockIdx.x * 16;
    const int h  = blockIdx.y;
    const int b  = blockIdx.z;
    const long headOff = ((long)(b * H + h)) * T * HS;
    const _Float16* qh = q + headOff;
    const _Float16* kh = k + headOff;
    const _Float16* vh = vT + headOff;  // [HS, T]

    v16h qa[2];
    qa[0] = load_a_frag(qh, HS, q0, 0);
    qa[1] = load_a_frag(qh, HS, q0, 32);

    v16h ones;
#pragma unroll
    for (int i = 0; i < 16; ++i) ones[i] = (_Float16)1.0f;

    float mi[8], li[8], alpha[8];
    v8f o[4] = {};
#pragma unroll
    for (int r = 0; r < 8; ++r) { mi[r] = -1e30f; li[r] = 0.0f; }

    for (int s0 = 0; s0 < q0 + 16; s0 += 32) {
        v8f sc[2] = {};
#pragma unroll
        for (int t = 0; t < 2; ++t)
#pragma unroll
            for (int kk = 0; kk < 2; ++kk) {
                v16h bf = load_b_frag_nk(kh, HS, s0 + t * 16, kk * 32);
                sc[t] = wmma_f16(qa[kk], bf, sc[t]);
            }

#pragma unroll
        for (int r = 0; r < 8; ++r) {
            const int query = q0 + sub * 8 + r;
            float v0 = sc[0][r] * scale;
            float v1 = sc[1][r] * scale;
            if (s0 + nn > query)      v0 = -1e30f;
            if (s0 + 16 + nn > query) v1 = -1e30f;
            float rm = fmaxf(v0, v1);
#pragma unroll
            for (int off = 1; off < 16; off <<= 1)
                rm = fmaxf(rm, __shfl_xor(rm, off, 32));
            const float mnew = fmaxf(mi[r], rm);
            alpha[r] = __expf(mi[r] - mnew);
            mi[r] = mnew;
            const int row = sub * 8 + r;
            pbuf[row * 32 + nn]      = (_Float16)__expf(v0 - mnew);
            pbuf[row * 32 + 16 + nn] = (_Float16)__expf(v1 - mnew);
        }
#pragma unroll
        for (int t = 0; t < 4; ++t)
#pragma unroll
            for (int r = 0; r < 8; ++r) o[t][r] *= alpha[r];

        v16h pa = load_a_frag(pbuf, 32, 0, 0);
        v8f zc = {};
        v8f rsv = wmma_f16(pa, ones, zc);   // all 16 row-sums in one WMMA
#pragma unroll
        for (int r = 0; r < 8; ++r) li[r] = li[r] * alpha[r] + rsv[r];

#pragma unroll
        for (int t = 0; t < 4; ++t) {
            v16h bv = load_b_frag_nk(vh, T, t * 16, s0);  // contiguous V^T
            o[t] = wmma_f16(pa, bv, o[t]);
        }
    }

#pragma unroll
    for (int t = 0; t < 4; ++t)
#pragma unroll
        for (int r = 0; r < 8; ++r) {
            const int row = q0 + sub * 8 + r;
            attn[((long)(b * T + row)) * C + h * HS + t * 16 + nn] =
                (_Float16)(o[t][r] / li[r]);
        }
}

// ---------------------------------------------------------------------------
// Launch
// ---------------------------------------------------------------------------
extern "C" void kernel_launch(void* const* d_in, const int* in_sizes, int n_in,
                              void* d_out, int out_size, void* d_ws, size_t ws_size,
                              hipStream_t stream) {
    constexpr int Bb = 2, T = 2048, Cc = 1024, H = 16, HS = 64, FF = 4096;
    constexpr long BT = (long)Bb * T;  // 4096

    const float* x     = (const float*)d_in[0];
    const float* Wq    = (const float*)d_in[1];
    const float* Wk    = (const float*)d_in[2];
    const float* Wv    = (const float*)d_in[3];
    const float* Wproj = (const float*)d_in[4];
    const float* W1    = (const float*)d_in[5];
    const float* b1    = (const float*)d_in[6];
    const float* W2    = (const float*)d_in[7];
    float* out = (float*)d_out;

    char* wsp = (char*)d_ws;
    auto alloc = [&](size_t bytes) -> char* {
        char* p = wsp;
        wsp += (bytes + 255) & ~(size_t)255;
        return p;
    };
    _Float16* xf   = (_Float16*)alloc(BT * Cc * 2);
    _Float16* Wq16 = (_Float16*)alloc((long)H * Cc * HS * 2);  // [H,HS,C]
    _Float16* Wk16 = (_Float16*)alloc((long)H * Cc * HS * 2);
    _Float16* Wv16 = (_Float16*)alloc((long)H * Cc * HS * 2);
    _Float16* Wp16 = (_Float16*)alloc((long)Cc * Cc * 2);
    _Float16* W116 = (_Float16*)alloc((long)FF * Cc * 2);
    _Float16* W216 = (_Float16*)alloc((long)Cc * FF * 2);
    _Float16* q16  = (_Float16*)alloc(BT * Cc * 2);            // [B,H,T,HS]
    _Float16* k16  = (_Float16*)alloc(BT * Cc * 2);            // [B,H,T,HS]
    _Float16* v16T = (_Float16*)alloc(BT * Cc * 2);            // [B,H,HS,T]
    _Float16* at16 = (_Float16*)alloc(BT * Cc * 2);
    float*    x1f  = (float*)   alloc(BT * Cc * 4);
    _Float16* x116 = (_Float16*)alloc(BT * Cc * 2);
    _Float16* h16  = (_Float16*)alloc(BT * (long)FF * 2);

    auto cvt = [&](const float* s, _Float16* d, long n) {
        cvt_f32_f16<<<(int)((n + 1023) / 1024), 256, 0, stream>>>(s, d, n);
    };
    auto cvtT = [&](const float* s, _Float16* d, int R, int Ccols, long n) {
        cvt_t_f32_f16<<<(int)((n + 1023) / 1024), 256, 0, stream>>>(s, d, R, Ccols, n);
    };
    cvt(x, xf, BT * Cc);
    cvtT(Wq, Wq16, Cc, HS, (long)H * Cc * HS);   // [H,C,HS] -> [H,HS,C]
    cvtT(Wk, Wk16, Cc, HS, (long)H * Cc * HS);
    cvtT(Wv, Wv16, Cc, HS, (long)H * Cc * HS);
    cvt(Wproj, Wp16, (long)Cc * Cc);             // already [N,K]
    cvt(W1, W116, (long)FF * Cc);
    cvt(W2, W216, (long)Cc * FF);

    // --- QKV projections: per (b,h): x[b][T,C] @ W[h][HS,C]^T -> [T,HS] ---
    {
        dim3 g(HS / 64, T / 128, Bb * H);  // (1, 16, 32)
        const long sA = (long)T * Cc, sB = (long)HS * Cc, sO = (long)T * HS;
        gemm16<false, false, false><<<g, 128, 0, stream>>>(
            xf, Wq16, nullptr, nullptr, nullptr, q16, T, HS, Cc, H, sA, H, sB, sO);
        gemm16<false, false, false><<<g, 128, 0, stream>>>(
            xf, Wk16, nullptr, nullptr, nullptr, k16, T, HS, Cc, H, sA, H, sB, sO);
        gemm16<false, false, true><<<g, 128, 0, stream>>>(   // V stored [HS,T]
            xf, Wv16, nullptr, nullptr, nullptr, v16T, T, HS, Cc, H, sA, H, sB, sO);
    }

    // --- causal flash attention -> at16 [B,T,C] ---
    attn_kernel<<<dim3(T / 16, H, Bb), 32, 0, stream>>>(
        q16, k16, v16T, at16, T, H, HS, Cc, 0.125f);

    // --- output proj + residual: x1 = x + attn @ Wproj^T ---
    gemm16<false, true, false><<<dim3(Cc / 64, BT / 128, 1), 128, 0, stream>>>(
        at16, Wp16, nullptr, x, x1f, x116, (int)BT, Cc, Cc, 1, 0, 1, 0, 0);

    // --- MLP1: h = silu(x1 @ W1^T + b1) ---
    gemm16<true, false, false><<<dim3(FF / 64, BT / 128, 1), 128, 0, stream>>>(
        x116, W116, b1, nullptr, nullptr, h16, (int)BT, FF, Cc, 1, 0, 1, 0, 0);

    // --- MLP2 + residual: out = x1 + h @ W2^T ---
    gemm16<false, true, false><<<dim3(Cc / 64, BT / 128, 1), 128, 0, stream>>>(
        h16, W216, nullptr, x1f, out, nullptr, (int)BT, Cc, FF, 1, 0, 1, 0, 0);
}
